// AttnDecoderRNN_20040317403351
// MI455X (gfx1250) — matrix operand order repaired and checked by reading, not compiled, not run on hardware
//
#include <hip/hip_runtime.h>
#include <hip/hip_bf16.h>

// ---------------------------------------------------------------------------
// AttnDecoderRNN for MI455X (gfx1250, wave32, WMMA).
// B=256, S=10, H=1024, V=32000, T=10.
// All heavy GEMMs run through v_wmma_f32_16x16x32_bf16 with fp32 accumulate.
// Weights converted to bf16 once (W_out bf16 = 65.5MB -> L2 resident across
// the 10 decode steps on the 192MB L2).
// ---------------------------------------------------------------------------

typedef __bf16 bf16_t;
typedef __attribute__((ext_vector_type(16))) __bf16 v16bf;
typedef __attribute__((ext_vector_type(8)))  __bf16 v8bf;
typedef __attribute__((ext_vector_type(8)))  float  v8f;

#define B_  256
#define S_  10
#define H_  1024
#define V_  32000
#define T_  10

// ---------------------------------------------------------------------------
// Generic bf16 GEMM:  C[m,n] = sum_k A[m,k] * Bm[n,k] + bias[n]
//   A:  [M,K] bf16 row-major,  Bm: [N,K] bf16 row-major (i.e. C = A * Bm^T)
//   C:  fp32, row stride ldc.
// Grid: grid.x = N/128, grid.y = M/16.  Block: 256 threads = 8 waves.
// Each wave owns one 16x16 tile; 8 waves share the same 16 A-rows.
// Fragment layouts per CDNA5 ISA 7.12.2 (16-bit A 16x32, B 32x16, f32 C/D).
// ---------------------------------------------------------------------------
__global__ __launch_bounds__(256)
void wmma_gemm_bias_kernel(const bf16_t* __restrict__ A,
                           const bf16_t* __restrict__ Bm,
                           const float*  __restrict__ bias,
                           float*        __restrict__ C,
                           int ldc, int K)
{
    const int lane   = threadIdx.x & 31;
    const int wave   = threadIdx.x >> 5;
    const int m0     = blockIdx.y << 4;
    const int n0     = (blockIdx.x << 7) + (wave << 4);
    const int half16 = lane >> 4;            // 0: lanes 0-15, 1: lanes 16-31

    const int arow = m0 + (lane & 15);
    const int ncol = n0 + (lane & 15);

    // A: lane holds row `arow`; element j (j=0..15) is K = (j>>3)*16 + 8*half16 + (j&7)
    const bf16_t* aptr = A  + (size_t)arow * K + (half16 << 3);
    // B: lane holds col `ncol`; element j is K = (j>>3)*16 + 8*half16 + (j&7)
    const bf16_t* bptr = Bm + (size_t)ncol * K + (half16 << 3);

    v8f acc = {};
    #pragma unroll 4
    for (int k0 = 0; k0 < K; k0 += 32) {
        // prefetch the streaming weight operand ahead (global_prefetch_b8)
        __builtin_prefetch(bptr + k0 + 512, 0, 1);

        v8bf a_lo = *(const v8bf*)(aptr + k0);        // K = kb+0..kb+7
        v8bf a_hi = *(const v8bf*)(aptr + k0 + 16);   // K = 16+kb+0..7
        v8bf b_lo = *(const v8bf*)(bptr + k0);        // K = kb..kb+7
        v8bf b_hi = *(const v8bf*)(bptr + k0 + 16);   // K = 16+kb+0..7

        v16bf a, b;
        #pragma unroll
        for (int j = 0; j < 8; ++j) {
            a[j]     = a_lo[j];
            a[j + 8] = a_hi[j];
            b[j]     = b_lo[j];
            b[j + 8] = b_hi[j];
        }
        acc = __builtin_amdgcn_wmma_f32_16x16x32_bf16(
                  /*neg_a=*/false, a, /*neg_b=*/false, b,
                  /*c_mod=*/(short)0, acc, /*reuse_a=*/false, /*reuse_b=*/false);
    }

    const float bn = bias ? bias[ncol] : 0.0f;
    // C/D layout: VGPR r -> M = m0 + 8*half16 + r, N = ncol
    float* cp = C + (size_t)(m0 + (half16 << 3)) * ldc + ncol;
    #pragma unroll
    for (int r = 0; r < 8; ++r)
        cp[(size_t)r * ldc] = acc[r] + bn;
}

// ---------------------------------------------------------------------------
// Elementwise helpers
// ---------------------------------------------------------------------------
__global__ void f32_to_bf16_kernel(const float* __restrict__ src,
                                   bf16_t* __restrict__ dst, long long n)
{
    long long i = (long long)blockIdx.x * 256 + threadIdx.x;
    if (i < n) dst[i] = (bf16_t)src[i];
}

__global__ void init_h_kernel(const float* __restrict__ eh,
                              float* __restrict__ h, bf16_t* __restrict__ h_bf)
{
    int i = blockIdx.x * 256 + threadIdx.x;    // B*H threads
    float v = eh[i];
    h[i] = v;
    h_bf[i] = (bf16_t)v;
}

// x = concat(emb[tok], ctx) in bf16, [B, 2H]
__global__ void build_x_kernel(const float* __restrict__ emb,
                               const int*   __restrict__ target,
                               int t,
                               const float* __restrict__ ctx,
                               bf16_t* __restrict__ x_bf)
{
    int idx = blockIdx.x * 256 + threadIdx.x;  // B*2H threads
    int b = idx >> 11;           // / 2048
    int k = idx & 2047;
    float v;
    if (k < H_) {
        int tok = (t == 0) ? 0 : target[b * T_ + (t - 1)];   // teacher forcing, SOS=0
        v = emb[(size_t)tok * H_ + k];
    } else {
        v = ctx[b * H_ + (k - H_)];
    }
    x_bf[idx] = (bf16_t)v;
}

__device__ __forceinline__ float sigmoidf_(float x) {
    return 1.0f / (1.0f + __expf(-x));
}

// GRU update from gi [B,3H], gh [B,3H]; h updated in place, h_bf refreshed.
__global__ void gru_kernel(const float* __restrict__ gi,
                           const float* __restrict__ gh,
                           float* __restrict__ h, bf16_t* __restrict__ h_bf)
{
    int i = blockIdx.x * 256 + threadIdx.x;    // B*H threads
    int b = i >> 10;
    int k = i & 1023;
    size_t base = (size_t)b * (3 * H_);
    float r  = sigmoidf_(gi[base + k]            + gh[base + k]);
    float z  = sigmoidf_(gi[base + H_ + k]       + gh[base + H_ + k]);
    float nn = tanhf    (gi[base + 2 * H_ + k] + r * gh[base + 2 * H_ + k]);
    float hn = (1.0f - z) * nn + z * h[i];
    h[i] = hn;
    h_bf[i] = (bf16_t)hn;
}

// ---------------------------------------------------------------------------
// Bahdanau attention: score -> softmax(S=10) -> ctx, one block per batch row.
// ---------------------------------------------------------------------------
__global__ __launch_bounds__(256)
void attn_kernel(const float* __restrict__ keysW,   // [B,S,H] = Wa(keys)+ba
                 const float* __restrict__ q,       // [B,H]   = Ua(h)+bua
                 const float* __restrict__ Va,      // [H]
                 const float* __restrict__ bva,     // [1]
                 const float* __restrict__ keys,    // [B,S,H]
                 float* __restrict__ ctx,           // [B,H]
                 float* __restrict__ attn_out,      // base of attentions block
                 int t)
{
    __shared__ float red[256];
    __shared__ float w[16];
    const int b = blockIdx.x, tid = threadIdx.x;
    const float* qb = q + (size_t)b * H_;

    for (int s = 0; s < S_; ++s) {
        const float* kw = keysW + ((size_t)(b * S_ + s)) * H_;
        float p = 0.0f;
        for (int hh = tid; hh < H_; hh += 256)
            p += tanhf(kw[hh] + qb[hh]) * Va[hh];
        red[tid] = p;
        __syncthreads();
        for (int st = 128; st > 0; st >>= 1) {
            if (tid < st) red[tid] += red[tid + st];
            __syncthreads();
        }
        if (tid == 0) w[s] = red[0] + bva[0];
        __syncthreads();
    }

    if (tid == 0) {
        float mx = w[0];
        for (int s = 1; s < S_; ++s) mx = fmaxf(mx, w[s]);
        float sum = 0.0f;
        for (int s = 0; s < S_; ++s) { w[s] = __expf(w[s] - mx); sum += w[s]; }
        float inv = 1.0f / sum;
        for (int s = 0; s < S_; ++s) w[s] *= inv;
    }
    __syncthreads();

    if (tid < S_)
        attn_out[(size_t)b * (T_ * S_) + t * S_ + tid] = w[tid];

    for (int hh = tid; hh < H_; hh += 256) {
        const float* kb = keys + ((size_t)b * S_) * H_ + hh;
        float acc = 0.0f;
        #pragma unroll
        for (int s = 0; s < S_; ++s) acc += w[s] * kb[(size_t)s * H_];
        ctx[(size_t)b * H_ + hh] = acc;
    }
}

// ---------------------------------------------------------------------------
// In-place log_softmax over V=32000; one block per (b,t) row of dec_out.
// ---------------------------------------------------------------------------
__global__ __launch_bounds__(256)
void log_softmax_kernel(float* __restrict__ out)
{
    __shared__ float red[256];
    __shared__ float bc;
    float* row = out + (size_t)blockIdx.x * V_;
    const int tid = threadIdx.x;

    float mx = -3.402823466e38f;
    for (int v = tid; v < V_; v += 256) mx = fmaxf(mx, row[v]);
    red[tid] = mx;
    __syncthreads();
    for (int st = 128; st > 0; st >>= 1) {
        if (tid < st) red[tid] = fmaxf(red[tid], red[tid + st]);
        __syncthreads();
    }
    if (tid == 0) bc = red[0];
    __syncthreads();
    const float smax = bc;
    __syncthreads();

    float sum = 0.0f;
    for (int v = tid; v < V_; v += 256) sum += __expf(row[v] - smax);
    red[tid] = sum;
    __syncthreads();
    for (int st = 128; st > 0; st >>= 1) {
        if (tid < st) red[tid] += red[tid + st];
        __syncthreads();
    }
    if (tid == 0) bc = __logf(red[0]);
    __syncthreads();
    const float lse = smax + bc;

    for (int v = tid; v < V_; v += 256) row[v] -= lse;
}

__global__ void copy_kernel(const float* __restrict__ src,
                            float* __restrict__ dst, int n)
{
    int i = blockIdx.x * 256 + threadIdx.x;
    if (i < n) dst[i] = src[i];
}

// ---------------------------------------------------------------------------
// Host-side orchestration
// ---------------------------------------------------------------------------
extern "C" void kernel_launch(void* const* d_in, const int* in_sizes, int n_in,
                              void* d_out, int out_size, void* d_ws, size_t ws_size,
                              hipStream_t stream)
{
    (void)in_sizes; (void)n_in; (void)out_size; (void)ws_size;

    const float* enc_out  = (const float*)d_in[0];   // [B,S,H]
    const float* enc_hid  = (const float*)d_in[1];   // [1,B,H]
    const int*   target   = (const int*)  d_in[2];   // [B,T]
    const float* emb      = (const float*)d_in[3];   // [V,H]
    const float* Wa       = (const float*)d_in[4];   // [H,H]
    const float* ba       = (const float*)d_in[5];   // [H]
    const float* Ua       = (const float*)d_in[6];   // [H,H]
    const float* bua      = (const float*)d_in[7];   // [H]
    const float* Va       = (const float*)d_in[8];   // [1,H]
    const float* bva      = (const float*)d_in[9];   // [1]
    const float* W_ih     = (const float*)d_in[10];  // [3H,2H]
    const float* W_hh     = (const float*)d_in[11];  // [3H,H]
    const float* b_ih     = (const float*)d_in[12];  // [3H]
    const float* b_hh     = (const float*)d_in[13];  // [3H]
    const float* W_out    = (const float*)d_in[14];  // [V,H]
    const float* b_out    = (const float*)d_in[15];  // [V]

    float* out = (float*)d_out;
    // output layout: dec_out [B,T,V] | h_last [1,B,H] | attentions [B,T*S,1]
    float* dec_out  = out;
    float* h_last   = out + (size_t)B_ * T_ * V_;
    float* attn_out = h_last + (size_t)B_ * H_;

    // ---- workspace carving (256B aligned) ----
    char*  ws  = (char*)d_ws;
    size_t off = 0;
    auto carve = [&](size_t bytes) -> char* {
        char* p = ws + off;
        off += (bytes + 255) & ~(size_t)255;
        return p;
    };
    bf16_t* Wout_bf = (bf16_t*)carve((size_t)V_ * H_ * 2);        // 65.5 MB
    bf16_t* Wih_bf  = (bf16_t*)carve((size_t)3 * H_ * 2 * H_ * 2);// 12.6 MB
    bf16_t* Whh_bf  = (bf16_t*)carve((size_t)3 * H_ * H_ * 2);    //  6.3 MB
    bf16_t* Ua_bf   = (bf16_t*)carve((size_t)H_ * H_ * 2);
    bf16_t* Wa_bf   = (bf16_t*)carve((size_t)H_ * H_ * 2);
    bf16_t* keys_bf = (bf16_t*)carve((size_t)B_ * S_ * H_ * 2);
    float*  keysW   = (float*) carve((size_t)B_ * S_ * H_ * 4);   // 10.5 MB
    float*  h       = (float*) carve((size_t)B_ * H_ * 4);
    bf16_t* h_bf    = (bf16_t*)carve((size_t)B_ * H_ * 2);
    float*  q       = (float*) carve((size_t)B_ * H_ * 4);
    float*  ctx     = (float*) carve((size_t)B_ * H_ * 4);
    bf16_t* x_bf    = (bf16_t*)carve((size_t)B_ * 2 * H_ * 2);
    float*  gi      = (float*) carve((size_t)B_ * 3 * H_ * 4);
    float*  gh      = (float*) carve((size_t)B_ * 3 * H_ * 4);

    auto cvt = [&](const float* s, bf16_t* d, long long n) {
        f32_to_bf16_kernel<<<(unsigned)((n + 255) / 256), 256, 0, stream>>>(s, d, n);
    };

    // ---- one-time precompute ----
    cvt(W_out, Wout_bf, (long long)V_ * H_);
    cvt(W_ih,  Wih_bf,  (long long)3 * H_ * 2 * H_);
    cvt(W_hh,  Whh_bf,  (long long)3 * H_ * H_);
    cvt(Ua,    Ua_bf,   (long long)H_ * H_);
    cvt(Wa,    Wa_bf,   (long long)H_ * H_);
    cvt(enc_out, keys_bf, (long long)B_ * S_ * H_);
    init_h_kernel<<<(B_ * H_) / 256, 256, 0, stream>>>(enc_hid, h, h_bf);

    // keysW[b,s,o] = keys @ Wa^T + ba : M = B*S = 2560, N = H, K = H
    wmma_gemm_bias_kernel<<<dim3(H_ / 128, (B_ * S_) / 16), 256, 0, stream>>>(
        keys_bf, Wa_bf, ba, keysW, H_, H_);

    // ---- decode loop (teacher forcing, T=10) ----
    for (int t = 0; t < T_; ++t) {
        // q = h @ Ua^T + bua : 256 x 1024, K=1024
        wmma_gemm_bias_kernel<<<dim3(H_ / 128, B_ / 16), 256, 0, stream>>>(
            h_bf, Ua_bf, bua, q, H_, H_);

        // attention: score->softmax->ctx  (+ writes attention weights output)
        attn_kernel<<<B_, 256, 0, stream>>>(keysW, q, Va, bva, enc_out,
                                            ctx, attn_out, t);

        // x = [emb[tok], ctx] bf16
        build_x_kernel<<<(B_ * 2 * H_) / 256, 256, 0, stream>>>(
            emb, target, t, ctx, x_bf);

        // gi = x @ W_ih^T + b_ih : 256 x 3072, K=2048
        wmma_gemm_bias_kernel<<<dim3((3 * H_) / 128, B_ / 16), 256, 0, stream>>>(
            x_bf, Wih_bf, b_ih, gi, 3 * H_, 2 * H_);
        // gh = h @ W_hh^T + b_hh : 256 x 3072, K=1024
        wmma_gemm_bias_kernel<<<dim3((3 * H_) / 128, B_ / 16), 256, 0, stream>>>(
            h_bf, Whh_bf, b_hh, gh, 3 * H_, H_);

        // GRU elementwise update (h, h_bf refreshed in place)
        gru_kernel<<<(B_ * H_) / 256, 256, 0, stream>>>(gi, gh, h, h_bf);

        // logits = h_new @ W_out^T + b_out, written straight into dec_out
        // row b, col v -> dec_out[(b*T + t)*V + v]  => base = dec_out + t*V, ldc = T*V
        wmma_gemm_bias_kernel<<<dim3(V_ / 128, B_ / 16), 256, 0, stream>>>(
            h_bf, Wout_bf, b_out, dec_out + (size_t)t * V_, T_ * V_, H_);
    }

    // log_softmax over V for all B*T rows, in place
    log_softmax_kernel<<<B_ * T_, 256, 0, stream>>>(dec_out);

    // h_last = final hidden state
    copy_kernel<<<(B_ * H_) / 256, 256, 0, stream>>>(h, h_last, B_ * H_);
}
